// Net_35940286333619
// MI455X (gfx1250) — matrix operand order repaired and checked by reading, not compiled, not run on hardware
//
#include <hip/hip_runtime.h>
#include <hip/hip_bf16.h>

typedef __attribute__((ext_vector_type(16))) _Float16 v16h;
typedef __attribute__((ext_vector_type(8)))  _Float16 v8h;
typedef __attribute__((ext_vector_type(8)))  float    v8f;

#define L_TOK 65536
#define DMOD  5
#define DIN   10
#define DST   16
#define DCONV 4
#define NBLK  8
#define NCH   256
#define CLEN  256

__device__ __forceinline__ float siluf(float x)     { return x / (1.f + __expf(-x)); }
__device__ __forceinline__ float softplusf(float x) { return (x > 20.f) ? x : log1pf(__expf(x)); }

// ---------- stage 0: build token stream x[l][c] from nearest-upsampled lms + pan ----------
__global__ void build_x_kernel(const float* __restrict__ lms, const float* __restrict__ pan,
                               float* __restrict__ x, float* __restrict__ resid) {
  int l = blockIdx.x * blockDim.x + threadIdx.x;
  if (l >= L_TOK) return;
  int h = l >> 8, w = l & 255;
#pragma unroll
  for (int c = 0; c < 4; ++c)
    x[l * DMOD + c] = lms[(c * 64 + (h >> 2)) * 64 + (w >> 2)];
  x[l * DMOD + 4] = pan[l];
#pragma unroll
  for (int c = 0; c < DMOD; ++c) resid[l * DMOD + c] = 0.f;
}

// ---------- residual add + layernorm (per token, DM=5) ----------
__global__ void resid_ln_kernel(const float* __restrict__ x, float* __restrict__ resid,
                                const float* __restrict__ g, const float* __restrict__ b,
                                float* __restrict__ xn) {
  int l = blockIdx.x * blockDim.x + threadIdx.x;
  if (l >= L_TOK) return;
  float v[DMOD]; float mu = 0.f;
#pragma unroll
  for (int c = 0; c < DMOD; ++c) {
    v[c] = resid[l * DMOD + c] + x[l * DMOD + c];
    resid[l * DMOD + c] = v[c];
    mu += v[c];
  }
  mu *= (1.f / DMOD);
  float var = 0.f;
#pragma unroll
  for (int c = 0; c < DMOD; ++c) { float d = v[c] - mu; var += d * d; }
  var *= (1.f / DMOD);
  float r = rsqrtf(var + 1e-5f);
#pragma unroll
  for (int c = 0; c < DMOD; ++c)
    xn[l * DMOD + c] = (v[c] - mu) * r * g[c] + b[c];
}

// ---------- input projection: (L,5) @ (20,5)^T -> (L,20) ----------
__global__ void in_proj_kernel(const float* __restrict__ xn, const float* __restrict__ w,
                               float* __restrict__ xz) {
  int l = blockIdx.x * blockDim.x + threadIdx.x;
  if (l >= L_TOK) return;
  float v[DMOD];
#pragma unroll
  for (int c = 0; c < DMOD; ++c) v[c] = xn[l * DMOD + c];
#pragma unroll
  for (int e = 0; e < 2 * DIN; ++e) {
    float s = 0.f;
#pragma unroll
    for (int d = 0; d < DMOD; ++d) s += v[d] * w[e * DMOD + d];
    xz[l * 2 * DIN + e] = s;
  }
}

// ---------- causal depthwise conv (k=4) + silu ----------
__global__ void dwconv_kernel(const float* __restrict__ xz, const float* __restrict__ w,
                              const float* __restrict__ b, float* __restrict__ xi) {
  int idx = blockIdx.x * blockDim.x + threadIdx.x;
  if (idx >= L_TOK * DIN) return;
  int l = idx / DIN, d = idx - l * DIN;
  float s = b[d];
#pragma unroll
  for (int k = 0; k < DCONV; ++k) {
    int lp = l - (DCONV - 1) + k;
    if (lp >= 0) s += w[d * DCONV + k] * xz[lp * 2 * DIN + d];
  }
  xi[idx] = siluf(s);
}

// ---------- x_dbl projection (10->33) and dt=softplus(dtr*dt_w+dt_b) ----------
__global__ void xproj_dt_kernel(const float* __restrict__ xi, const float* __restrict__ xw,
                                const float* __restrict__ dtw, const float* __restrict__ dtb,
                                float* __restrict__ Bm, float* __restrict__ Cm,
                                float* __restrict__ dt) {
  int l = blockIdx.x * blockDim.x + threadIdx.x;
  if (l >= L_TOK) return;
  float v[DIN];
#pragma unroll
  for (int d = 0; d < DIN; ++d) v[d] = xi[l * DIN + d];
  float dtr = 0.f;
#pragma unroll
  for (int e = 0; e < 1 + 2 * DST; ++e) {
    float s = 0.f;
#pragma unroll
    for (int d = 0; d < DIN; ++d) s += v[d] * xw[e * DIN + d];
    if (e == 0)            dtr = s;
    else if (e < 1 + DST)  Bm[l * DST + (e - 1)] = s;
    else                   Cm[l * DST + (e - 1 - DST)] = s;
  }
#pragma unroll
  for (int d = 0; d < DIN; ++d)
    dt[l * DIN + d] = softplusf(dtr * dtw[d] + dtb[d]);
}

// ---------- SSM scan pass 1: per-(chunk,d) aggregates over 16 states ----------
__global__ void scan_p1_kernel(const float* __restrict__ dt, const float* __restrict__ xi,
                               const float* __restrict__ Bm, const float* __restrict__ Alog,
                               float* __restrict__ Aagg, float* __restrict__ Bagg) {
  int t = blockIdx.x * blockDim.x + threadIdx.x;
  if (t >= NCH * DIN) return;
  int c = t / DIN, d = t - c * DIN;
  float A[DST], aP[DST], bC[DST];
#pragma unroll
  for (int n = 0; n < DST; ++n) {
    A[n] = -__expf(Alog[d * DST + n]);
    aP[n] = 1.f; bC[n] = 0.f;
  }
  for (int s = 0; s < CLEN; ++s) {
    int l = c * CLEN + s;
    float dtl = dt[l * DIN + d];
    float u   = dtl * xi[l * DIN + d];
#pragma unroll
    for (int n = 0; n < DST; ++n) {
      float a = __expf(dtl * A[n]);
      bC[n] = a * bC[n] + u * Bm[l * DST + n];
      aP[n] *= a;
    }
  }
#pragma unroll
  for (int n = 0; n < DST; ++n) {
    Aagg[t * DST + n] = aP[n];
    Bagg[t * DST + n] = bC[n];
  }
}

// ---------- SSM scan pass 2: exclusive prefix over chunk aggregates ----------
__global__ void scan_p2_kernel(const float* __restrict__ Aagg, const float* __restrict__ Bagg,
                               float* __restrict__ Hcar) {
  int t = threadIdx.x;
  if (t >= DIN * DST) return;
  int d = t / DST, n = t - d * DST;
  float h = 0.f;
  for (int c = 0; c < NCH; ++c) {
    int idx = (c * DIN + d) * DST + n;
    Hcar[idx] = h;
    h = Aagg[idx] * h + Bagg[idx];
  }
}

// ---------- SSM scan pass 3: replay with carry, y = (h.C + xi*Dp) * silu(z) ----------
__global__ void scan_p3_kernel(const float* __restrict__ dt, const float* __restrict__ xi,
                               const float* __restrict__ Bm, const float* __restrict__ Cm,
                               const float* __restrict__ xz, const float* __restrict__ Alog,
                               const float* __restrict__ Dp, const float* __restrict__ Hcar,
                               float* __restrict__ y) {
  int t = blockIdx.x * blockDim.x + threadIdx.x;
  if (t >= NCH * DIN) return;
  int c = t / DIN, d = t - c * DIN;
  float A[DST], hn[DST];
#pragma unroll
  for (int n = 0; n < DST; ++n) {
    A[n]  = -__expf(Alog[d * DST + n]);
    hn[n] = Hcar[(c * DIN + d) * DST + n];
  }
  float Dpd = Dp[d];
  for (int s = 0; s < CLEN; ++s) {
    int l = c * CLEN + s;
    float dtl = dt[l * DIN + d];
    float u   = dtl * xi[l * DIN + d];
    float acc = 0.f;
#pragma unroll
    for (int n = 0; n < DST; ++n) {
      float a = __expf(dtl * A[n]);
      hn[n] = a * hn[n] + u * Bm[l * DST + n];
      acc  += hn[n] * Cm[l * DST + n];
    }
    float z = xz[l * 2 * DIN + DIN + d];
    y[l * DIN + d] = (acc + xi[l * DIN + d] * Dpd) * siluf(z);
  }
}

// ---------- output projection: (L,10) @ (5,10)^T -> (L,5) ----------
__global__ void out_proj_kernel(const float* __restrict__ y, const float* __restrict__ w,
                                float* __restrict__ x) {
  int l = blockIdx.x * blockDim.x + threadIdx.x;
  if (l >= L_TOK) return;
  float v[DIN];
#pragma unroll
  for (int d = 0; d < DIN; ++d) v[d] = y[l * DIN + d];
#pragma unroll
  for (int e = 0; e < DMOD; ++e) {
    float s = 0.f;
#pragma unroll
    for (int d = 0; d < DIN; ++d) s += v[d] * w[e * DIN + d];
    x[l * DMOD + e] = s;
  }
}

// ---------- pack token-major x into padded HWC f16 (265 rows x 264 x 16ch; row 264 = guard) ----------
__global__ void pack_input_kernel(const float* __restrict__ x, _Float16* __restrict__ xh) {
  int idx = blockIdx.x * blockDim.x + threadIdx.x;
  const int total = 265 * 264 * 16;
  if (idx >= total) return;
  int c = idx & 15;
  int p = idx >> 4;
  int hp = p / 264, wp = p - hp * 264;
  float v = 0.f;
  if (c < 5 && hp >= 4 && hp < 260 && wp >= 4 && wp < 260)
    v = x[((hp - 4) * 256 + (wp - 4)) * DMOD + c];
  xh[idx] = (_Float16)v;
}

__global__ void fill_h_kernel(_Float16* __restrict__ p, int n) {
  int i = blockIdx.x * blockDim.x + threadIdx.x;
  if (i < n) p[i] = (_Float16)0.f;
}

// ---------- pack conv weights to f16 [Npad][Kpad], K ordered (kh, kw, cin-padded) ----------
__global__ void pack_w_hwc_kernel(const float* __restrict__ w, _Float16* __restrict__ wh,
                                  int nreal, int npad, int cin_real, int cinp,
                                  int kwsz, int kpad) {
  int idx = blockIdx.x * blockDim.x + threadIdx.x;
  if (idx >= npad * kpad) return;
  int n  = idx / kpad;
  int kp = idx - n * kpad;
  int cell = kp / cinp;
  int cin  = kp - cell * cinp;
  int kh = cell / kwsz;
  int kw = cell - kh * kwsz;
  float v = 0.f;
  if (n < nreal && cell < kwsz * kwsz && cin < cin_real)
    v = w[((n * cin_real + cin) * kwsz + kh) * kwsz + kw];
  wh[idx] = (_Float16)v;
}

// ---------- implicit-GEMM conv via V_WMMA_F32_16X16X32_F16 ----------
// A = weights, LDS-staged once per block (all 4 waves share the same 16-channel tile)
//     -> two contiguous ds_load_b128 per lane per chunk
// B = activations (32 K x 16 pixels, HWC) -> one contiguous v16h global load per lane per chunk
template<int KW, int CINP, int KPAD, int INW, int OUTPAD, int OUTW, int COUTP,
         int NREAL, bool RELU, bool TOF16>
__global__ __launch_bounds__(128)
void conv_wmma_kernel(const _Float16* __restrict__ xin,
                      const _Float16* __restrict__ wh,
                      const float* __restrict__ bias,
                      _Float16* __restrict__ outh,
                      float* __restrict__ outf) {
  __shared__ _Float16 smem[16 * KPAD];
  int gid  = blockIdx.x * blockDim.x + threadIdx.x;
  int wvid = gid >> 5;
  int lane = threadIdx.x & 31;
  const int MT = 4096;                   // 16-pixel tiles, row-aligned
  int mt = wvid % MT, nt = wvid / MT;    // blocks of 4 waves never straddle nt

  // cooperative stage of the 16 x KPAD weight tile into LDS (vectorized, 16B chunks)
  {
    const v8h* src = (const v8h*)(wh + (size_t)nt * 16 * KPAD);
    v8h* dst = (v8h*)smem;
    const int nvec = 16 * KPAD / 8;
    for (int t = threadIdx.x; t < nvec; t += 128) dst[t] = src[t];
  }
  __syncthreads();

  int pb = mt * 16;
  int nloc = lane & 15;
  int pix = pb + nloc;
  int h = pix >> 8, w = pix & 255;
  const _Float16* bbase = xin + (h * INW + w) * CINP;
  int s0   = (lane & 16) ? 16 : 0;       // A frag K offset (contiguous, matches B)
  int boff = (lane & 16) ? 16 : 0;       // B frag K offset
  const _Float16* arow = smem + nloc * KPAD;
  v8f acc = {};
#pragma unroll
  for (int kb = 0; kb < KPAD; kb += 32) {
    v8h wa0 = *(const v8h*)(arow + kb + s0);        // halves 0..7 : K = kb+s0..+7
    v8h wa1 = *(const v8h*)(arow + kb + s0 + 8);    // halves 8..15: K = kb+s0+8..+15
    v16h a;
#pragma unroll
    for (int i = 0; i < 8; ++i) { a[i] = wa0[i]; a[i + 8] = wa1[i]; }
    int k0   = kb + boff;                // 16 consecutive K per lane, cin-innermost
    int cell = k0 / CINP;
    int c0   = k0 - cell * CINP;
    int kh   = cell / KW;
    int kw2  = cell - kh * KW;
    v16h b = *(const v16h*)(bbase + (kh * INW + kw2) * CINP + c0);
    if (kb + 32 < KPAD) {                // prefetch next chunk's activation line
      int k1 = kb + 32 + boff;
      int cell1 = k1 / CINP;
      int c1 = k1 - cell1 * CINP;
      int kh1 = cell1 / KW;
      int kw1 = cell1 - kh1 * KW;
      __builtin_prefetch(bbase + (kh1 * INW + kw1) * CINP + c1, 0, 3);
    }
    acc = __builtin_amdgcn_wmma_f32_16x16x32_f16(false, a, false, b, (short)0, acc,
                                                 false, false);
  }
  int cbase = nt * 16 + ((lane & 16) ? 8 : 0);
  if (TOF16) {
    v8h ov;
#pragma unroll
    for (int r = 0; r < 8; ++r) {
      float v = acc[r] + bias[cbase + r];
      if (RELU) v = fmaxf(v, 0.f);
      ov[r] = (_Float16)v;
    }
    *(v8h*)(outh + ((h + OUTPAD) * OUTW + (w + OUTPAD)) * COUTP + cbase) = ov;
  } else {
#pragma unroll
    for (int r = 0; r < 8; ++r) {
      int c = cbase + r;
      if (c < NREAL) {
        float v = acc[r] + bias[c];
        if (RELU) v = fmaxf(v, 0.f);
        outf[(c * 256 + h) * 256 + w] = v;
      }
    }
  }
}

extern "C" void kernel_launch(void* const* d_in, const int* in_sizes, int n_in,
                              void* d_out, int out_size, void* d_ws, size_t ws_size,
                              hipStream_t stream) {
  (void)in_sizes; (void)n_in; (void)out_size; (void)ws_size;
  const float* lms    = (const float*)d_in[0];
  /* d_in[1] = mms is unused by the reference */
  const float* pan    = (const float*)d_in[2];
  const float* norm_g = (const float*)d_in[3];
  const float* norm_b = (const float*)d_in[4];
  const float* in_w   = (const float*)d_in[5];
  const float* conv_w = (const float*)d_in[6];
  const float* conv_b = (const float*)d_in[7];
  const float* xproj_w= (const float*)d_in[8];
  const float* dt_w   = (const float*)d_in[9];
  const float* dt_b   = (const float*)d_in[10];
  const float* A_log  = (const float*)d_in[11];
  const float* Dp     = (const float*)d_in[12];
  const float* out_w  = (const float*)d_in[13];
  const float* c1w    = (const float*)d_in[14];
  const float* c1b    = (const float*)d_in[15];
  const float* c2w    = (const float*)d_in[16];
  const float* c2b    = (const float*)d_in[17];
  const float* c3w    = (const float*)d_in[18];
  const float* c3b    = (const float*)d_in[19];
  float* outp = (float*)d_out;

  char* ws = (char*)d_ws;
  size_t off = 0;
  auto alloc = [&](size_t bytes) -> void* {
    void* p = (void*)(ws + off);
    off = (off + bytes + 255) & ~(size_t)255;
    return p;
  };
  float* x     = (float*)alloc((size_t)L_TOK * DMOD * 4);
  float* resid = (float*)alloc((size_t)L_TOK * DMOD * 4);
  float* xn    = (float*)alloc((size_t)L_TOK * DMOD * 4);
  float* xz    = (float*)alloc((size_t)L_TOK * 2 * DIN * 4);
  float* xi    = (float*)alloc((size_t)L_TOK * DIN * 4);
  float* Bm    = (float*)alloc((size_t)L_TOK * DST * 4);
  float* Cm    = (float*)alloc((size_t)L_TOK * DST * 4);
  float* dt    = (float*)alloc((size_t)L_TOK * DIN * 4);
  float* Aagg  = (float*)alloc((size_t)NCH * DIN * DST * 4);
  float* Bagg  = (float*)alloc((size_t)NCH * DIN * DST * 4);
  float* Hcar  = (float*)alloc((size_t)NCH * DIN * DST * 4);
  float* y     = (float*)alloc((size_t)L_TOK * DIN * 4);
  _Float16* xh1 = (_Float16*)alloc((size_t)265 * 264 * 16 * 2);  // HWC, 16ch, +guard row
  _Float16* xh2 = (_Float16*)alloc((size_t)260 * 260 * 64 * 2);  // HWC
  _Float16* xh3 = (_Float16*)alloc((size_t)260 * 260 * 32 * 2);  // HWC
  _Float16* w1h = (_Float16*)alloc((size_t)64 * 1312 * 2);
  _Float16* w2h = (_Float16*)alloc((size_t)32 * 1600 * 2);
  _Float16* w3h = (_Float16*)alloc((size_t)16 * 800  * 2);

  build_x_kernel<<<L_TOK / 256, 256, 0, stream>>>(lms, pan, x, resid);

  for (int i = 0; i < NBLK; ++i) {
    resid_ln_kernel<<<L_TOK / 256, 256, 0, stream>>>(
        x, resid, norm_g + i * DMOD, norm_b + i * DMOD, xn);
    in_proj_kernel<<<L_TOK / 256, 256, 0, stream>>>(
        xn, in_w + (size_t)i * 2 * DIN * DMOD, xz);
    dwconv_kernel<<<(L_TOK * DIN + 255) / 256, 256, 0, stream>>>(
        xz, conv_w + (size_t)i * DIN * DCONV, conv_b + (size_t)i * DIN, xi);
    xproj_dt_kernel<<<L_TOK / 256, 256, 0, stream>>>(
        xi, xproj_w + (size_t)i * (1 + 2 * DST) * DIN,
        dt_w + (size_t)i * DIN, dt_b + (size_t)i * DIN, Bm, Cm, dt);
    scan_p1_kernel<<<(NCH * DIN + 127) / 128, 128, 0, stream>>>(
        dt, xi, Bm, A_log + (size_t)i * DIN * DST, Aagg, Bagg);
    scan_p2_kernel<<<1, DIN * DST, 0, stream>>>(Aagg, Bagg, Hcar);
    scan_p3_kernel<<<(NCH * DIN + 127) / 128, 128, 0, stream>>>(
        dt, xi, Bm, Cm, xz, A_log + (size_t)i * DIN * DST,
        Dp + (size_t)i * DIN, Hcar, y);
    out_proj_kernel<<<L_TOK / 256, 256, 0, stream>>>(
        y, out_w + (size_t)i * DMOD * DIN, x);
  }

  pack_input_kernel<<<(265 * 264 * 16 + 255) / 256, 256, 0, stream>>>(x, xh1);
  fill_h_kernel<<<(260 * 260 * 64 + 255) / 256, 256, 0, stream>>>(xh2, 260 * 260 * 64);
  fill_h_kernel<<<(260 * 260 * 32 + 255) / 256, 256, 0, stream>>>(xh3, 260 * 260 * 32);
  pack_w_hwc_kernel<<<(64 * 1312 + 255) / 256, 256, 0, stream>>>(c1w, w1h, 64, 64, 5, 16, 9, 1312);
  pack_w_hwc_kernel<<<(32 * 1600 + 255) / 256, 256, 0, stream>>>(c2w, w2h, 32, 32, 64, 64, 5, 1600);
  pack_w_hwc_kernel<<<(16 * 800  + 255) / 256, 256, 0, stream>>>(c3w, w3h, 4, 16, 32, 32, 5, 800);

  // conv1: 9x9, 5(pad16)->64, pad 4. waves = 4096*4 -> 4096 blocks of 128.
  conv_wmma_kernel<9, 16, 1312, 264, 2, 260, 64, 64, true, true>
      <<<4096, 128, 0, stream>>>(xh1, w1h, c1b, xh2, nullptr);
  // conv2: 5x5, 64->32, pad 2. waves = 4096*2 -> 2048 blocks.
  conv_wmma_kernel<5, 64, 1600, 260, 2, 260, 32, 32, true, true>
      <<<2048, 128, 0, stream>>>(xh2, w2h, c2b, xh3, nullptr);
  // conv3: 5x5, 32->4 (padded to 16), pad 2, float out, no relu.
  conv_wmma_kernel<5, 32, 800, 260, 0, 256, 16, 4, false, false>
      <<<1024, 128, 0, stream>>>(xh3, w3h, c3b, nullptr, outp);
}
